// ScaledDotProductAttention_20942260535739
// MI455X (gfx1250) — matrix operand order repaired and checked by reading, not compiled, not run on hardware
//
#include <hip/hip_runtime.h>
#include <hip/hip_bf16.h>

#define BS_ 8
#define S_  512
#define DK_ 2048

typedef __attribute__((ext_vector_type(16))) __bf16 v16bf;
typedef __attribute__((ext_vector_type(8)))  __bf16 v8bf;
typedef __attribute__((ext_vector_type(8)))  float  v8f;

__device__ __forceinline__ unsigned short f32_to_bf16_rne(float f) {
    unsigned int u = __float_as_uint(f);
    unsigned int r = u + 0x7FFFu + ((u >> 16) & 1u);
    return (unsigned short)(r >> 16);
}

// Build one 16-element bf16 fragment from two 16B chunks (global or LDS pointer).
__device__ __forceinline__ v16bf load_frag(const __bf16* p0, const __bf16* p1) {
    v8bf lo = *(const v8bf*)p0;
    v8bf hi = *(const v8bf*)p1;
    return __builtin_shufflevector(lo, hi, 0,1,2,3,4,5,6,7,8,9,10,11,12,13,14,15);
}

__device__ __forceinline__ v8f wmma_bf16(v16bf a, v16bf b, v8f c) {
    return __builtin_amdgcn_wmma_f32_16x16x32_bf16(
        false, a, false, b, (short)0, c, false, false);
}

// Async global -> LDS, 16B per lane (512B per instruction). ASYNCcnt tracked.
__device__ __forceinline__ void async_b128(unsigned lds_off, unsigned long long gaddr) {
    asm volatile("global_load_async_to_lds_b128 %0, %1, off"
                 :: "v"(lds_off), "v"(gaddr) : "memory");
}
__device__ __forceinline__ void wait_async_16() { asm volatile("s_wait_asynccnt 0x10" ::: "memory"); }
__device__ __forceinline__ void wait_async_12() { asm volatile("s_wait_asynccnt 0xc"  ::: "memory"); }
__device__ __forceinline__ void wait_async_0()  { asm volatile("s_wait_asynccnt 0x0"  ::: "memory"); }

// ---------------- Phase 0a: transpose [b][rows][cols] f32 -> [b][cols][rows] bf16
__global__ __launch_bounds__(256)
void transpose_to_bf16(const float* __restrict__ in, unsigned short* __restrict__ out,
                       int rows, int cols) {
    __shared__ float tile[32][33];
    int b  = blockIdx.z;
    int c0 = blockIdx.x * 32;
    int r0 = blockIdx.y * 32;
    const float* inb = in + (size_t)b * rows * cols;
    unsigned short* outb = out + (size_t)b * rows * cols;
    int tx = threadIdx.x;
    int ty = threadIdx.y;
    #pragma unroll
    for (int i = ty; i < 32; i += 8)
        tile[i][tx] = inb[(size_t)(r0 + i) * cols + c0 + tx];
    __syncthreads();
    #pragma unroll
    for (int i = ty; i < 32; i += 8)
        outb[(size_t)(c0 + i) * rows + r0 + tx] = f32_to_bf16_rne(tile[tx][i]);
}

// ---------------- Phase 0b: straight f32 -> bf16 convert (v)
__global__ __launch_bounds__(256)
void convert_to_bf16(const float* __restrict__ in, unsigned short* __restrict__ out,
                     unsigned int n4) {
    unsigned int i = blockIdx.x * blockDim.x + threadIdx.x;
    if (i >= n4) return;
    float4 f = ((const float4*)in)[i];
    unsigned int lo = (unsigned int)f32_to_bf16_rne(f.x) | ((unsigned int)f32_to_bf16_rne(f.y) << 16);
    unsigned int hi = (unsigned int)f32_to_bf16_rne(f.z) | ((unsigned int)f32_to_bf16_rne(f.w) << 16);
    ((uint2*)out)[i] = make_uint2(lo, hi);
}

// ---------------- Phase 1: scores = scale * qT@k, softmax over d, attn bf16 [b][d][e]
// 16 waves; wave w owns d in [w*128, w*128+128) = 8 WMMA tiles.
// All 16 B fragments (whole K) preloaded into registers (128 VGPRs), pinned above
// the async asm. A panel staged per-wave via async LDS, double buffered, no barriers.
// LDS: 16 waves * 2 bufs * (128 rows * 80B) = 320 KiB; softmax scratch aliases it after.
__global__ __launch_bounds__(512)
void attn_scores_softmax(const unsigned short* __restrict__ qT_,  // [b][DK][S] bf16
                         const unsigned short* __restrict__ kT_,  // [b][DK][S] bf16
                         unsigned short* __restrict__ attn_)      // [b][DK][DK] bf16
{
    __shared__ __align__(16) unsigned char smem[16 * 2 * 128 * 80];   // 327680 B

    int bid  = blockIdx.x;
    int b    = bid >> 7;
    int e0   = (bid & 127) << 4;
    int wave = threadIdx.x >> 5;      // 0..15
    int lane = threadIdx.x & 31;
    int n    = lane & 15;
    int hl   = lane >> 4;
    int dwave = wave << 7;            // wave * 128

    const __bf16* qT = (const __bf16*)qT_ + (size_t)b * DK_ * S_;
    const __bf16* kT = (const __bf16*)kT_ + (size_t)b * DK_ * S_;
    const __bf16* krow = kT + (size_t)(e0 + n) * S_;

    // Preload the ENTIRE B panel (16 K-steps) into registers; the asm "memory"
    // clobbers below prevent these loads from sinking into the loop.
    v16bf bfr[16];
    #pragma unroll
    for (int j = 0; j < 16; ++j) {
        const __bf16* p = krow + j * 32 + hl * 8;
        bfr[j] = load_frag(p, p + 16);
    }

    // Async staging setup: slice rows are d-rows of qT (1024B apart in global, 80B in LDS).
    unsigned smemBase = (unsigned)(size_t)&smem[0];
    unsigned waveBase = (unsigned)(wave * 20480);
    unsigned wrLane   = ((unsigned)(lane >> 2)) * 80u + ((unsigned)(lane & 3)) * 16u;
    unsigned long long gRow =
        (unsigned long long)(size_t)(qT + (size_t)(dwave + (lane >> 2)) * S_)
        + (unsigned long long)((lane & 3) * 16);

    auto issue_slice = [&](int s0, int buf) {
        unsigned long long g = gRow + (unsigned long long)(s0 * 2);
        unsigned l = smemBase + waveBase + (unsigned)(buf * 10240) + wrLane;
        #pragma unroll
        for (int j = 0; j < 16; ++j) {        // 16 x 512B = 8KB slice (128 rows x 64B)
            async_b128(l, g);
            g += 8192;                        // 8 rows * 1024B
            l += 640;                         // 8 rows * 80B
        }
    };

    v8f acc[8];
    v8f zero = {0.f,0.f,0.f,0.f,0.f,0.f,0.f,0.f};
    #pragma unroll
    for (int t = 0; t < 8; ++t) acc[t] = zero;

    issue_slice(0, 0);
    #pragma unroll
    for (int step = 0; step < 16; ++step) {
        int buf = step & 1;
        if (step < 15) {
            issue_slice((step + 1) * 32, buf ^ 1);
            wait_async_16();                  // this step's slice complete, next in flight
        } else {
            wait_async_0();
        }
        const unsigned char* base = &smem[wave * 20480 + buf * 10240];
        #pragma unroll
        for (int t = 0; t < 8; ++t) {
            const __bf16* p = (const __bf16*)(base + (t * 16 + n) * 80 + hl * 16);
            v16bf a = load_frag(p, p + 16);   // +32 bytes
            acc[t] = wmma_bf16(a, bfr[step], acc[t]);
        }
    }

    __syncthreads();                          // buffers dead; alias softmax scratch
    float* red = (float*)smem;                // [16][16]

    const float scale = 0.022097086912079612f;   // 1/sqrt(2048)
    float m = -3.0e38f;
    #pragma unroll
    for (int t = 0; t < 8; ++t)
        #pragma unroll
        for (int r = 0; r < 8; ++r) {
            float x = acc[t][r] * scale;
            acc[t][r] = x;
            m = fmaxf(m, x);
        }
    m = fmaxf(m, __shfl_xor(m, 16, 32));
    if (lane < 16) red[wave * 16 + lane] = m;
    __syncthreads();
    float gm = red[n];
    #pragma unroll
    for (int w = 1; w < 16; ++w) gm = fmaxf(gm, red[w * 16 + n]);
    __syncthreads();

    float ssum = 0.f;
    #pragma unroll
    for (int t = 0; t < 8; ++t)
        #pragma unroll
        for (int r = 0; r < 8; ++r) {
            float ex = __expf(acc[t][r] - gm);
            acc[t][r] = ex;
            ssum += ex;
        }
    ssum += __shfl_xor(ssum, 16, 32);
    if (lane < 16) red[wave * 16 + lane] = ssum;
    __syncthreads();
    float gs = 0.f;
    #pragma unroll
    for (int w = 0; w < 16; ++w) gs += red[w * 16 + n];
    float inv = 1.0f / gs;

    unsigned short* ab = attn_ + (size_t)b * DK_ * DK_;
    #pragma unroll
    for (int t = 0; t < 8; ++t) {
        int dbase = dwave + t * 16 + hl * 8;
        #pragma unroll
        for (int r = 0; r < 8; ++r) {
            int d = dbase + r;
            ab[(size_t)d * DK_ + e0 + n] = f32_to_bf16_rne(acc[t][r] * inv);
        }
    }
}

// ---------------- Phase 2: out[b][s][d] = sum_e attn[d][e] * v[s][e]   (f32 out)
// Workgroup: 128(s) x 128(d), 8 waves; wave: 32(s) x 64(d) = 2x4 WMMA tiles.
// Per-wave async LDS staging of A(v) and B(attn) slices, double buffered.
// LDS: 8 waves * 2 bufs * (32+64 rows * 80B) = 120 KiB.
__global__ __launch_bounds__(256)
void attn_out_gemm(const unsigned short* __restrict__ vbf_,   // [b][S][DK] bf16
                   const unsigned short* __restrict__ attn_,  // [b][DK][DK] bf16
                   float* __restrict__ out)                   // [b][S][DK] f32
{
    __shared__ __align__(16) unsigned char smem[8 * 2 * 96 * 80];     // 122880 B

    int b    = blockIdx.z;
    int sblk = blockIdx.y * 128;
    int dblk = blockIdx.x * 128;
    int wave = threadIdx.x >> 5;
    int lane = threadIdx.x & 31;
    int n    = lane & 15;
    int hl   = lane >> 4;
    int wy = wave & 3;
    int wx = wave >> 2;
    int sbase = sblk + wy * 32;
    int dbase = dblk + wx * 64;

    const __bf16* vb = (const __bf16*)vbf_  + (size_t)b * S_  * DK_;
    const __bf16* at = (const __bf16*)attn_ + (size_t)b * DK_ * DK_;

    unsigned smemBase = (unsigned)(size_t)&smem[0];
    unsigned waveBase = (unsigned)(wave * 15360);          // 2 bufs * 7680
    unsigned wrLane   = ((unsigned)(lane >> 2)) * 80u + ((unsigned)(lane & 3)) * 16u;
    unsigned long long gA =
        (unsigned long long)(size_t)(vb + (size_t)(sbase + (lane >> 2)) * DK_)
        + (unsigned long long)((lane & 3) * 16);
    unsigned long long gB =
        (unsigned long long)(size_t)(at + (size_t)(dbase + (lane >> 2)) * DK_)
        + (unsigned long long)((lane & 3) * 16);

    auto issue_step = [&](int e0, int buf) {
        unsigned lb = smemBase + waveBase + (unsigned)(buf * 7680);
        unsigned long long g = gA + (unsigned long long)(e0 * 2);
        unsigned l = lb + wrLane;
        #pragma unroll
        for (int j = 0; j < 4; ++j) {          // A: 32 s-rows
            async_b128(l, g);
            g += 32768;                        // 8 rows * 4096B
            l += 640;
        }
        g = gB + (unsigned long long)(e0 * 2);
        l = lb + 2560u + wrLane;
        #pragma unroll
        for (int j = 0; j < 8; ++j) {          // B: 64 d-rows
            async_b128(l, g);
            g += 32768;
            l += 640;
        }
    };

    v8f acc[2][4];
    v8f zero = {0.f,0.f,0.f,0.f,0.f,0.f,0.f,0.f};
    #pragma unroll
    for (int mt = 0; mt < 2; ++mt)
        #pragma unroll
        for (int nt = 0; nt < 4; ++nt) acc[mt][nt] = zero;

    issue_step(0, 0);
    for (int e0 = 0; e0 < DK_; e0 += 32) {
        int buf = (e0 >> 5) & 1;
        if (e0 + 32 < DK_) {
            issue_step(e0 + 32, buf ^ 1);
            __builtin_prefetch(vb + (size_t)(sbase + n) * DK_ + ((e0 + 256) & (DK_ - 1)), 0, 2);
            __builtin_prefetch(at + (size_t)(dbase + n) * DK_ + ((e0 + 256) & (DK_ - 1)), 0, 2);
            wait_async_12();                   // step e0 complete, e0+32 in flight
        } else {
            wait_async_0();
        }
        const unsigned char* base  = &smem[wave * 15360 + buf * 7680];
        const unsigned char* bbase = base + 2560;
        v16bf afr[2];
        #pragma unroll
        for (int mt = 0; mt < 2; ++mt) {
            const __bf16* p = (const __bf16*)(base + (mt * 16 + n) * 80 + hl * 16);
            afr[mt] = load_frag(p, p + 16);    // +32 bytes
        }
        #pragma unroll
        for (int nt = 0; nt < 4; ++nt) {
            const __bf16* p = (const __bf16*)(bbase + (nt * 16 + n) * 80 + hl * 16);
            v16bf bfr = load_frag(p, p + 16);   // +32 bytes
            acc[0][nt] = wmma_bf16(afr[0], bfr, acc[0][nt]);
            acc[1][nt] = wmma_bf16(afr[1], bfr, acc[1][nt]);
        }
    }

    float* ob = out + (size_t)b * S_ * DK_;
    #pragma unroll
    for (int mt = 0; mt < 2; ++mt)
        #pragma unroll
        for (int r = 0; r < 8; ++r) {
            int s = sbase + mt * 16 + hl * 8 + r;
            #pragma unroll
            for (int nt = 0; nt < 4; ++nt) {
                int d = dbase + nt * 16 + n;
                ob[(size_t)s * DK_ + d] = acc[mt][nt][r];
            }
        }
}

extern "C" void kernel_launch(void* const* d_in, const int* in_sizes, int n_in,
                              void* d_out, int out_size, void* d_ws, size_t ws_size,
                              hipStream_t stream) {
    const float* q = (const float*)d_in[0];
    const float* k = (const float*)d_in[1];
    const float* v = (const float*)d_in[2];
    float* out = (float*)d_out;

    const size_t SZ_QKV = (size_t)BS_ * S_ * DK_ * sizeof(unsigned short); // 16 MiB each
    unsigned char* ws = (unsigned char*)d_ws;
    unsigned short* qT   = (unsigned short*)(ws);                 // [b][DK][S]
    unsigned short* kT   = (unsigned short*)(ws + SZ_QKV);        // [b][DK][S]
    unsigned short* vbf  = (unsigned short*)(ws + 2 * SZ_QKV);    // [b][S][DK]
    unsigned short* attn = (unsigned short*)(ws + 3 * SZ_QKV);    // [b][DK][DK] = 64 MiB

    dim3 tb(32, 8);
    transpose_to_bf16<<<dim3(DK_ / 32, S_ / 32, BS_), tb, 0, stream>>>(q, qT, S_, DK_);
    transpose_to_bf16<<<dim3(DK_ / 32, S_ / 32, BS_), tb, 0, stream>>>(k, kT, S_, DK_);
    unsigned int n4 = (unsigned int)((size_t)BS_ * S_ * DK_ / 4);
    convert_to_bf16<<<(n4 + 255) / 256, 256, 0, stream>>>(v, vbf, n4);

    attn_scores_softmax<<<BS_ * (DK_ / 16), 512, 0, stream>>>(qT, kT, attn);
    attn_out_gemm<<<dim3(DK_ / 128, S_ / 128, BS_), 256, 0, stream>>>(vbf, attn, out);
}
